// MLP_68101001446017
// MI455X (gfx1250) — compile-verified, matrix operand order checked by reading, not run on hardware
//
#include <hip/hip_runtime.h>
#include <hip/hip_bf16.h>
#include <math.h>

// ---------------------------------------------------------------------------
// GNN edge predictor for MI455X (gfx1250), wave32 + v_wmma_f32_16x16x32_bf16,
// with weights staged into LDS via gfx1250 async-to-LDS loads (ASYNCcnt).
//
//   [convert]  fold BN -> scale/shift; swizzle weights into WMMA B-fragment
//              layout (bf16) in workspace (edge W0 zero-padded K=16 -> 32).
//   [node]     per block: async-stage nW0+nW1 fragments (64KB) into LDS,
//              per wave: 16 nodes, h0 = relu(x@nW0+b0)  (4x8 WMMA),
//              h1 = relu(h0@nW1+b1) (4x8 WMMA), BN affine, then
//              sRow[n] = h1.pW[0:128], sCol[n] = h1.pW[128:256].
//   [edge]     per block: async-stage eW0+eW1 fragments (40KB) into LDS,
//              per wave: 16 edges, h0 (1x8 WMMA, K padded), h1 (4x8 WMMA),
//              BN affine, sE = h1.pW[256:384],
//              out[e] = sigmoid(sE + sRow[row[e]] + sCol[col[e]] + pb).
// ef ([E,128]) is never materialized; gathers are 4 bytes per endpoint.
// ---------------------------------------------------------------------------

typedef __bf16 bf16_t;
typedef __attribute__((ext_vector_type(16))) __bf16 v16bf;
typedef __attribute__((ext_vector_type(8)))  float  v8f;

#define GN_N       50000
#define GN_E       800000
#define GN_H       128
#define GN_DNODE   128
#define GN_DEDGE   16
#define GN_EPS     1e-5f
#define LDS_LD     136      // 128 + 8 bf16 pad: row stride 272B (16B aligned,
                            // bank rotation 4/row -> conflict-free b128 reads)

#define H_ELEMS    (8 * 16 * LDS_LD)             // h tiles, 8 waves
#define H_BYTES    (H_ELEMS * 2)                 // 34816 B
#define WB_EDGE    (5 * 4096 * 2)                // eW0(1kt) + eW1(4kt) frags
#define WB_NODE    (8 * 4096 * 2)                // nW0(4kt) + nW1(4kt) frags

// ---- A fragment (16x32 bf16) lane/K mapping (ISA 7.12.2, 16-bit A 16x32) ---
// lane<16 : M=lane,    K = {k0+0..7   (j=0..7),  k0+16..23 (j=8..15)}
// lane>=16: M=lane-16, K = {k0+8..15  (j=0..7),  k0+24..31 (j=8..15)}
__device__ __forceinline__ int afrag_k(int lane, int k0, int j) {
    return k0 + ((lane >> 4) ? 8 : 0) + ((j < 8) ? j : j + 8);
}

__device__ __forceinline__ v16bf a_frag_global_f32(const float* __restrict__ rowp,
                                                   int lane, int k0, int kvalid) {
    v16bf a;
#pragma unroll
    for (int j = 0; j < 16; ++j) {
        int k = afrag_k(lane, k0, j);
        float v = (k < kvalid) ? rowp[k] : 0.0f;
        a[j] = (bf16_t)v;
    }
    return a;
}

__device__ __forceinline__ v16bf a_frag_lds(const bf16_t* __restrict__ tile,
                                            int lane, int k0) {
    const bf16_t* p = tile + (lane & 15) * LDS_LD;
    v16bf a;
#pragma unroll
    for (int j = 0; j < 16; ++j) a[j] = p[afrag_k(lane, k0, j)];
    return a;
}

// ---- gfx1250 async copy: 16B global -> LDS per lane, tracked by ASYNCcnt ---
__device__ __forceinline__ void async_b128_to_lds(unsigned lds_off, int glb_off,
                                                  const void* gbase) {
    asm volatile("global_load_async_to_lds_b128 %0, %1, %2"
                 :: "v"(lds_off), "v"(glb_off), "s"(gbase) : "memory");
}
__device__ __forceinline__ void wait_async_zero() {
    asm volatile("s_wait_asynccnt 0x0" ::: "memory");
}

// Stage BYTES of pre-swizzled weight fragments into LDS (whole block).
template <int BYTES>
__device__ __forceinline__ void stage_weights(bf16_t* lds_dst, const bf16_t* gsrc) {
    unsigned lbase = (unsigned)(size_t)lds_dst;   // flat addr low 32 = LDS offset
    constexpr int CHUNKS = BYTES / 16;            // b128 chunks
    int tid = threadIdx.x;
#pragma unroll
    for (int i = 0; i < CHUNKS / 256; ++i) {
        int c = tid + i * 256;
        async_b128_to_lds(lbase + c * 16, c * 16, gsrc);
    }
}

// ---------------------------------------------------------------------------
// Weight swizzle: W[K][128] row-major f32 -> B fragments, bf16.
// dst[((kt*8+nt)*32 + lane)*16 + j]; K set mirrors the A layout. k>=Kvalid -> 0.
// ---------------------------------------------------------------------------
__global__ void gnn_convert_weights(const float* __restrict__ W, int Kvalid,
                                    int ktiles, bf16_t* __restrict__ dst) {
    int t = blockIdx.x * blockDim.x + threadIdx.x;
    int total = ktiles * 8 * 32 * 16;
    if (t >= total) return;
    int j    = t & 15;
    int lane = (t >> 4) & 31;
    int nt   = (t >> 9) & 7;
    int kt   = t >> 12;
    int n = nt * 16 + (lane & 15);
    int k = kt * 32 + afrag_k(lane, 0, j);
    float v = (k < Kvalid) ? W[(size_t)k * GN_H + n] : 0.0f;
    dst[t] = (bf16_t)v;
}

__global__ void gnn_bn_fold(const float* nw, const float* nb, const float* nm, const float* nv,
                            const float* ew, const float* eb, const float* em, const float* ev,
                            float* nscale, float* nshift, float* escale, float* eshift) {
    int t = threadIdx.x;
    if (t < GN_H) {
        float s = nw[t] * rsqrtf(nv[t] + GN_EPS);
        nscale[t] = s; nshift[t] = nb[t] - nm[t] * s;
    } else if (t < 2 * GN_H) {
        int c = t - GN_H;
        float s = ew[c] * rsqrtf(ev[c] + GN_EPS);
        escale[c] = s; eshift[c] = eb[c] - em[c] * s;
    }
}

// ---------------------------------------------------------------------------
// Node kernel: 8 waves/block, 16 nodes/wave. Dynamic LDS:
//   [0, WB_NODE)                  nW0+nW1 fragments (async-staged)
//   [WB_NODE, +H_BYTES)           h0 tiles (bf16, padded)
//   [... , +8*2*16*4)             reduction staging
// ---------------------------------------------------------------------------
__global__ __launch_bounds__(256) void gnn_node_kernel(
    const float* __restrict__ x,
    const bf16_t* __restrict__ Wg,        // nW0 frags then nW1 frags (contig)
    const float* __restrict__ b0, const float* __restrict__ b1,
    const float* __restrict__ scale, const float* __restrict__ shift,
    const float* __restrict__ pW,
    float* __restrict__ sRow, float* __restrict__ sCol) {
    extern __shared__ __align__(32) char smem[];
    bf16_t* wfrag = (bf16_t*)smem;
    bf16_t* hbuf  = (bf16_t*)(smem + WB_NODE);
    float*  stage = (float*)(smem + WB_NODE + H_BYTES);

    const int wave = threadIdx.x >> 5;
    const int lane = threadIdx.x & 31;
    const int half = lane >> 4;
    const int m    = lane & 15;
    const int base = (blockIdx.x * 8 + wave) * 16;
    bf16_t* hw = hbuf + wave * 16 * LDS_LD;

    // kick off async weight staging, overlap with A-fragment global loads
    stage_weights<WB_NODE>(wfrag, Wg);

    int row = base + m; if (row >= GN_N) row = GN_N - 1;   // clamp, uniform flow
    const float* xrow = x + (size_t)row * GN_DNODE;
    v16bf a0[4];
#pragma unroll
    for (int kt = 0; kt < 4; ++kt) a0[kt] = a_frag_global_f32(xrow, lane, kt * 32, GN_DNODE);

    wait_async_zero();
    __syncthreads();

    const v16bf* B0 = (const v16bf*)wfrag;            // 4 kt x 8 nt
    const v16bf* B1 = (const v16bf*)(wfrag + 4 * 4096);

    // ---- layer 0 ----
#pragma unroll
    for (int nt = 0; nt < 8; ++nt) {
        float bias = b0[nt * 16 + m];
        v8f acc;
#pragma unroll
        for (int r = 0; r < 8; ++r) acc[r] = bias;
#pragma unroll
        for (int kt = 0; kt < 4; ++kt) {
            v16bf b = B0[(kt * 8 + nt) * 32 + lane];
            acc = __builtin_amdgcn_wmma_f32_16x16x32_bf16(false, a0[kt], false, b,
                                                          (short)0, acc, false, false);
        }
#pragma unroll
        for (int r = 0; r < 8; ++r)
            hw[(r + 8 * half) * LDS_LD + nt * 16 + m] = (bf16_t)fmaxf(acc[r], 0.0f);
    }
    __syncthreads();

    // ---- layer 1 + BN + dual dot with pW[0:128], pW[128:256] ----
    v16bf a1[4];
#pragma unroll
    for (int kt = 0; kt < 4; ++kt) a1[kt] = a_frag_lds(hw, lane, kt * 32);

    float pr[8], pc[8];
#pragma unroll
    for (int r = 0; r < 8; ++r) { pr[r] = 0.0f; pc[r] = 0.0f; }

#pragma unroll
    for (int nt = 0; nt < 8; ++nt) {
        int   col  = nt * 16 + m;
        float bias = b1[col];
        v8f acc;
#pragma unroll
        for (int r = 0; r < 8; ++r) acc[r] = bias;
#pragma unroll
        for (int kt = 0; kt < 4; ++kt) {
            v16bf b = B1[(kt * 8 + nt) * 32 + lane];
            acc = __builtin_amdgcn_wmma_f32_16x16x32_bf16(false, a1[kt], false, b,
                                                          (short)0, acc, false, false);
        }
        float sc = scale[col], sh = shift[col];
        float wr = pW[col], wc = pW[GN_H + col];
#pragma unroll
        for (int r = 0; r < 8; ++r) {
            float v = fmaxf(acc[r], 0.0f) * sc + sh;   // relu then BN affine
            pr[r] += v * wr;
            pc[r] += v * wc;
        }
    }
#pragma unroll
    for (int r = 0; r < 8; ++r) {
#pragma unroll
        for (int d = 1; d < 16; d <<= 1) {
            pr[r] += __shfl_xor(pr[r], d);
            pc[r] += __shfl_xor(pc[r], d);
        }
    }
    if (m == 0) {                 // lanes 0 (rows 0..7) and 16 (rows 8..15)
#pragma unroll
        for (int r = 0; r < 8; ++r) {
            stage[(wave * 2 + 0) * 16 + r + 8 * half] = pr[r];
            stage[(wave * 2 + 1) * 16 + r + 8 * half] = pc[r];
        }
    }
    __syncthreads();
    if (lane < 16) {
        int node = base + lane;
        if (node < GN_N) {
            sRow[node] = stage[(wave * 2 + 0) * 16 + lane];
            sCol[node] = stage[(wave * 2 + 1) * 16 + lane];
        }
    }
}

// ---------------------------------------------------------------------------
// Edge kernel: 8 waves/block, 16 edges/wave. Fully fused MLP + gather + sigmoid.
// ---------------------------------------------------------------------------
__global__ __launch_bounds__(256) void gnn_edge_kernel(
    const float* __restrict__ ea, const int* __restrict__ eidx,
    const bf16_t* __restrict__ Wg,        // eW0 frags then eW1 frags (contig)
    const float* __restrict__ b0, const float* __restrict__ b1,
    const float* __restrict__ scale, const float* __restrict__ shift,
    const float* __restrict__ pW, const float* __restrict__ pb,
    const float* __restrict__ sRow, const float* __restrict__ sCol,
    float* __restrict__ out) {
    extern __shared__ __align__(32) char smem[];
    bf16_t* wfrag = (bf16_t*)smem;
    bf16_t* hbuf  = (bf16_t*)(smem + WB_EDGE);
    float*  stage = (float*)(smem + WB_EDGE + H_BYTES);

    const int wave = threadIdx.x >> 5;
    const int lane = threadIdx.x & 31;
    const int half = lane >> 4;
    const int m    = lane & 15;
    const int base = (blockIdx.x * 8 + wave) * 16;
    bf16_t* hw = hbuf + wave * 16 * LDS_LD;

    stage_weights<WB_EDGE>(wfrag, Wg);

    int row = base + m; if (row >= GN_E) row = GN_E - 1;
    const float* arow = ea + (size_t)row * GN_DEDGE;
    v16bf a0 = a_frag_global_f32(arow, lane, 0, GN_DEDGE);   // K=16 pad -> 32

    wait_async_zero();
    __syncthreads();

    const v16bf* B0 = (const v16bf*)wfrag;            // 1 kt x 8 nt
    const v16bf* B1 = (const v16bf*)(wfrag + 4096);

    // ---- layer 0 ----
#pragma unroll
    for (int nt = 0; nt < 8; ++nt) {
        float bias = b0[nt * 16 + m];
        v8f acc;
#pragma unroll
        for (int r = 0; r < 8; ++r) acc[r] = bias;
        v16bf b = B0[nt * 32 + lane];
        acc = __builtin_amdgcn_wmma_f32_16x16x32_bf16(false, a0, false, b,
                                                      (short)0, acc, false, false);
#pragma unroll
        for (int r = 0; r < 8; ++r)
            hw[(r + 8 * half) * LDS_LD + nt * 16 + m] = (bf16_t)fmaxf(acc[r], 0.0f);
    }
    __syncthreads();

    // ---- layer 1 + BN + dot with pW[256:384] ----
    v16bf a1[4];
#pragma unroll
    for (int kt = 0; kt < 4; ++kt) a1[kt] = a_frag_lds(hw, lane, kt * 32);

    float pe[8];
#pragma unroll
    for (int r = 0; r < 8; ++r) pe[r] = 0.0f;

#pragma unroll
    for (int nt = 0; nt < 8; ++nt) {
        int   col  = nt * 16 + m;
        float bias = b1[col];
        v8f acc;
#pragma unroll
        for (int r = 0; r < 8; ++r) acc[r] = bias;
#pragma unroll
        for (int kt = 0; kt < 4; ++kt) {
            v16bf b = B1[(kt * 8 + nt) * 32 + lane];
            acc = __builtin_amdgcn_wmma_f32_16x16x32_bf16(false, a1[kt], false, b,
                                                          (short)0, acc, false, false);
        }
        float sc = scale[col], sh = shift[col], we = pW[2 * GN_H + col];
#pragma unroll
        for (int r = 0; r < 8; ++r)
            pe[r] += (fmaxf(acc[r], 0.0f) * sc + sh) * we;
    }
#pragma unroll
    for (int r = 0; r < 8; ++r)
#pragma unroll
        for (int d = 1; d < 16; d <<= 1) pe[r] += __shfl_xor(pe[r], d);

    if (m == 0) {
#pragma unroll
        for (int r = 0; r < 8; ++r) stage[wave * 16 + r + 8 * half] = pe[r];
    }
    __syncthreads();

    if (lane < 16) {
        int e = base + lane;
        if (e < GN_E) {
            int   rn = eidx[e];
            int   cn = eidx[GN_E + e];
            float v  = stage[wave * 16 + lane] + sRow[rn] + sCol[cn] + pb[0];
            out[e]   = 1.0f / (1.0f + expf(-v));
        }
    }
}

// ---------------------------------------------------------------------------
extern "C" void kernel_launch(void* const* d_in, const int* in_sizes, int n_in,
                              void* d_out, int out_size, void* d_ws, size_t ws_size,
                              hipStream_t stream) {
    const float* x    = (const float*)d_in[0];
    const float* ea   = (const float*)d_in[1];
    const int*   eidx = (const int*)  d_in[2];
    const float* nW0  = (const float*)d_in[3];
    const float* nb0  = (const float*)d_in[4];
    const float* nW1  = (const float*)d_in[5];
    const float* nb1  = (const float*)d_in[6];
    const float* eW0  = (const float*)d_in[7];
    const float* eb0  = (const float*)d_in[8];
    const float* eW1  = (const float*)d_in[9];
    const float* eb1  = (const float*)d_in[10];
    const float* bnw  = (const float*)d_in[11];
    const float* bnb  = (const float*)d_in[12];
    const float* bnm  = (const float*)d_in[13];
    const float* bnv  = (const float*)d_in[14];
    const float* bew  = (const float*)d_in[15];
    const float* beb  = (const float*)d_in[16];
    const float* bem  = (const float*)d_in[17];
    const float* bev  = (const float*)d_in[18];
    const float* pW   = (const float*)d_in[19];
    const float* pb   = (const float*)d_in[20];
    float* out        = (float*)d_out;

    // Workspace layout. W0/W1 pairs are contiguous so one async stage covers
    // both layers of a kernel. All sizes are multiples of 256B.
    char* ws = (char*)d_ws;
    size_t off = 0;
    auto take = [&](size_t bytes) { char* p = ws + off; off += (bytes + 255) & ~(size_t)255; return p; };
    bf16_t* wsnW0 = (bf16_t*)take(4 * 4096 * sizeof(bf16_t));   // node W0 (4 kt)
    bf16_t* wsnW1 = (bf16_t*)take(4 * 4096 * sizeof(bf16_t));   // node W1 (4 kt)
    bf16_t* wseW0 = (bf16_t*)take(1 * 4096 * sizeof(bf16_t));   // edge W0 (K pad)
    bf16_t* wseW1 = (bf16_t*)take(4 * 4096 * sizeof(bf16_t));   // edge W1 (4 kt)
    float*  nsc   = (float*)take(GN_H * sizeof(float));
    float*  nsh   = (float*)take(GN_H * sizeof(float));
    float*  esc   = (float*)take(GN_H * sizeof(float));
    float*  esh   = (float*)take(GN_H * sizeof(float));
    float*  sRow  = (float*)take(GN_N * sizeof(float));
    float*  sCol  = (float*)take(GN_N * sizeof(float));
    (void)ws_size; (void)n_in; (void)in_sizes; (void)out_size;

    // 1) prep: swizzle weights to WMMA B-fragments, fold BN
    gnn_convert_weights<<<(4 * 4096 + 255) / 256, 256, 0, stream>>>(nW0, GN_DNODE, 4, wsnW0);
    gnn_convert_weights<<<(4 * 4096 + 255) / 256, 256, 0, stream>>>(nW1, GN_H,     4, wsnW1);
    gnn_convert_weights<<<(1 * 4096 + 255) / 256, 256, 0, stream>>>(eW0, GN_DEDGE, 1, wseW0);
    gnn_convert_weights<<<(4 * 4096 + 255) / 256, 256, 0, stream>>>(eW1, GN_H,     4, wseW1);
    gnn_bn_fold<<<1, 256, 0, stream>>>(bnw, bnb, bnm, bnv, bew, beb, bem, bev,
                                       nsc, nsh, esc, esh);

    // 2) node MLP -> per-node prediction scalars
    size_t smem_node = WB_NODE + H_BYTES + 8 * 2 * 16 * sizeof(float);   // ~99 KB
    gnn_node_kernel<<<(GN_N + 127) / 128, 256, smem_node, stream>>>(
        x, wsnW0, nb0, nb1, nsc, nsh, pW, sRow, sCol);

    // 3) fused edge MLP + gather + sigmoid
    size_t smem_edge = WB_EDGE + H_BYTES + 8 * 16 * sizeof(float);       // ~74 KB
    gnn_edge_kernel<<<(GN_E + 127) / 128, 256, smem_edge, stream>>>(
        ea, eidx, wseW0, eb0, eb1, esc, esh, pW, pb, sRow, sCol, out);
}
